// EdgeConv_8564164788720
// MI455X (gfx1250) — matrix operand order repaired and checked by reading, not compiled, not run on hardware
//
#include <hip/hip_runtime.h>
#include <stdint.h>

#define NNODES 20000
#define KNB 32
#define CIN 128
#define HDIM 256
#define NGROUPS (NNODES / 2)   // 2 nodes per block-iteration

typedef __attribute__((ext_vector_type(16))) __bf16 bf16x16;
typedef __attribute__((ext_vector_type(8)))  float  f32x8;

union OpPack {
    uint4   q[2];
    bf16x16 v;
};

// pack two fp32 -> two bf16 in one dword: +0x8000 round, byte-perm pack.
__device__ __forceinline__ uint32_t pk_bf16(float lo, float hi) {
    uint32_t a = __builtin_bit_cast(uint32_t, lo) + 0x8000u;
    uint32_t b = __builtin_bit_cast(uint32_t, hi) + 0x8000u;
    return __builtin_amdgcn_perm(b, a, 0x07060302u);  // {b[3],b[2],a[3],a[2]}
}
// single fp32 -> bf16 bits in low 16
__device__ __forceinline__ uint32_t bf16w(float f) {
    return (__builtin_bit_cast(uint32_t, f) + 0x8000u) >> 16;
}
__device__ __forceinline__ uint4 cvt8(float4 f0, float4 f1) {
    uint4 r;
    r.x = pk_bf16(f0.x, f0.y);
    r.y = pk_bf16(f0.z, f0.w);
    r.z = pk_bf16(f1.x, f1.y);
    r.w = pk_bf16(f1.z, f1.w);
    return r;
}
__device__ __forceinline__ uint4 cvt8sub(float4 n0, float4 v0, float4 n1, float4 v1) {
    uint4 r;
    r.x = pk_bf16(n0.x - v0.x, n0.y - v0.y);
    r.y = pk_bf16(n0.z - v0.z, n0.w - v0.w);
    r.z = pk_bf16(n1.x - v1.x, n1.y - v1.y);
    r.w = pk_bf16(n1.z - v1.z, n1.w - v1.w);
    return r;
}

// ---------------------------------------------------------------------------
// Kernel 1: convert W1/W2 (fp32 [256x256]) to bf16, pre-swizzled into the
// WMMA B-operand lane layout for 16x16x32 (8 contiguous dwords per lane per
// tile -> B operand load = 2x ds_load_b128, tile stride = 1KB).
// ---------------------------------------------------------------------------
__global__ void prep_weights_k(const float* __restrict__ W1,
                               const float* __restrict__ W2,
                               uint32_t* __restrict__ ws) {
    int d = blockIdx.x * blockDim.x + threadIdx.x;
    if (d >= 32768) return;
    int v    = d & 7;
    int c    = d >> 3;
    int lane = c & 31;
    int t    = c >> 5;
    int nt   = t & 15;
    int ks   = t >> 4;
    int n    = nt * 16 + (lane & 15);
    int k0   = ks * 32 + (lane >> 4) * 8 + (v >> 2) * 16 + 2 * (v & 3);
    ws[d]         = pk_bf16(W1[k0 * HDIM + n], W1[(k0 + 1) * HDIM + n]);
    ws[32768 + d] = pk_bf16(W2[k0 * HDIM + n], W2[(k0 + 1) * HDIM + n]);
}

// ---------------------------------------------------------------------------
// Kernel 2: fused EdgeConv. Block = 256 threads = 8 waves, 2 nodes per group.
// LDS: W1 (128KB) + W2 (128KB) swizzled bf16 + h1 staging (32KB) = 288KB.
// wave w: node_local = w&1, col-quarter nq = w>>1 (64 cols each).
// GEMM1 exploits that the x_v half of the edge feature is row-independent:
// its contribution is computed once (accV) and shared by both M-tiles.
// ---------------------------------------------------------------------------
__launch_bounds__(256)
__global__ void edgeconv_main_k(const float* __restrict__ X,
                                const int*  __restrict__ senders,
                                const float* __restrict__ b1,
                                const float* __restrict__ b2,
                                const uint32_t* __restrict__ wsw,
                                float* __restrict__ out) {
    __shared__ __align__(16) uint32_t s_w1[32768];        // 128 KB
    __shared__ __align__(16) uint32_t s_w2[32768];        // 128 KB
    __shared__ __align__(16) uint16_t s_h1[64 * HDIM];    // 32 KB (bf16)

    const int tid        = threadIdx.x;
    const int lane       = tid & 31;
    const int wave       = tid >> 5;
    const int node_local = wave & 1;
    const int nq         = wave >> 1;    // 0..3
    const int l16        = lane & 15;
    const int lh         = lane >> 4;    // 0/1

    // ---- load pre-swizzled weights into LDS (once per block) ----
    {
        const uint4* g = reinterpret_cast<const uint4*>(wsw);
        uint4* l1 = reinterpret_cast<uint4*>(s_w1);
        uint4* l2 = reinterpret_cast<uint4*>(s_w2);
        for (int i = tid; i < 8192; i += 256) {
            l1[i] = g[i];
            l2[i] = g[8192 + i];
        }
    }
    __syncthreads();

    // biases for this wave's 4 column tiles (invariant across groups)
    float bia1[4], bia2[4];
#pragma unroll
    for (int nt2 = 0; nt2 < 4; ++nt2) {
        int col   = nq * 64 + nt2 * 16 + l16;
        bia1[nt2] = b1[col];
        bia2[nt2] = b2[col];
    }

    for (int g = blockIdx.x; g < NGROUPS; g += gridDim.x) {
        const int gnode = g * 2 + node_local;

        const int   s0  = senders[gnode * KNB + l16];
        const int   s1  = senders[gnode * KNB + 16 + l16];
        const float* pV  = X + (size_t)gnode * CIN;
        const float* pS0 = X + (size_t)s0 * CIN;
        const float* pS1 = X + (size_t)s1 * CIN;

        // ================= GEMM1 (gather fused per k-step) =================
        f32x8 accV[4]    = {};   // shared x_v contribution
        f32x8 accD[2][4] = {};   // per-M-tile (x_n - x_v) contribution
#pragma unroll
        for (int j = 0; j < 4; ++j) {
            const int c0 = j * 32 + lh * 8;
            const int c1 = c0 + 16;
            // gather fp32 chunks (x_v reused for both converts)
            float4 v00 = *reinterpret_cast<const float4*>(pV + c0);
            float4 v01 = *reinterpret_cast<const float4*>(pV + c0 + 4);
            float4 v10 = *reinterpret_cast<const float4*>(pV + c1);
            float4 v11 = *reinterpret_cast<const float4*>(pV + c1 + 4);
            float4 p00 = *reinterpret_cast<const float4*>(pS0 + c0);
            float4 p01 = *reinterpret_cast<const float4*>(pS0 + c0 + 4);
            float4 p10 = *reinterpret_cast<const float4*>(pS0 + c1);
            float4 p11 = *reinterpret_cast<const float4*>(pS0 + c1 + 4);
            float4 q00 = *reinterpret_cast<const float4*>(pS1 + c0);
            float4 q01 = *reinterpret_cast<const float4*>(pS1 + c0 + 4);
            float4 q10 = *reinterpret_cast<const float4*>(pS1 + c1);
            float4 q11 = *reinterpret_cast<const float4*>(pS1 + c1 + 4);

            OpPack aV, aD0, aD1;
            aV.q[0]  = cvt8(v00, v01);
            aV.q[1]  = cvt8(v10, v11);
            aD0.q[0] = cvt8sub(p00, v00, p01, v01);
            aD0.q[1] = cvt8sub(p10, v10, p11, v11);
            aD1.q[0] = cvt8sub(q00, v00, q01, v01);
            aD1.q[1] = cvt8sub(q10, v10, q11, v11);

            // x_v part: k-step = j, one WMMA per column tile (shared M-tiles)
            {
                const uint32_t* base = &s_w1[((j * 16 + nq * 4) * 32 + lane) * 8];
                OpPack bx[4];
#pragma unroll
                for (int nt2 = 0; nt2 < 4; ++nt2) {
                    const uint4* pb = reinterpret_cast<const uint4*>(base + nt2 * 256);
                    bx[nt2].q[0] = pb[0];
                    bx[nt2].q[1] = pb[1];
                }
#pragma unroll
                for (int nt2 = 0; nt2 < 4; ++nt2)
                    accV[nt2] = __builtin_amdgcn_wmma_f32_16x16x32_bf16(
                        false, aV.v, false, bx[nt2].v, (short)0, accV[nt2], false, false);
            }
            // diff part: k-step = j + 4
            {
                const uint32_t* base = &s_w1[(((j + 4) * 16 + nq * 4) * 32 + lane) * 8];
                OpPack bd[4];
#pragma unroll
                for (int nt2 = 0; nt2 < 4; ++nt2) {
                    const uint4* pb = reinterpret_cast<const uint4*>(base + nt2 * 256);
                    bd[nt2].q[0] = pb[0];
                    bd[nt2].q[1] = pb[1];
                }
#pragma unroll
                for (int nt2 = 0; nt2 < 4; ++nt2) {
                    accD[0][nt2] = __builtin_amdgcn_wmma_f32_16x16x32_bf16(
                        false, aD0.v, false, bd[nt2].v, (short)0, accD[0][nt2], false, false);
                    accD[1][nt2] = __builtin_amdgcn_wmma_f32_16x16x32_bf16(
                        false, aD1.v, false, bd[nt2].v, (short)0, accD[1][nt2], false, false);
                }
            }
        }

        // epilogue: combine shared+diff, bias, relu -> bf16 h1 (plain b16 stores)
#pragma unroll
        for (int mt = 0; mt < 2; ++mt) {
#pragma unroll
            for (int nt2 = 0; nt2 < 4; ++nt2) {
                const int col = nq * 64 + nt2 * 16 + l16;
#pragma unroll
                for (int r = 0; r < 8; ++r) {
                    const int row = node_local * 32 + mt * 16 + r + 8 * lh;
                    float hv = accV[nt2][r] + accD[mt][nt2][r] + bia1[nt2];
                    hv = fmaxf(hv, 0.f);
                    s_h1[row * HDIM + col] = (uint16_t)bf16w(hv);
                }
            }
        }
        __syncthreads();

        // ================= GEMM2: h2 = h1 @ W2 + b2 =================
        f32x8 acc2[2][4] = {};
#pragma unroll
        for (int ks = 0; ks < 8; ++ks) {
            const int c0 = ks * 32 + lh * 8;
            OpPack a[2];
#pragma unroll
            for (int mt = 0; mt < 2; ++mt) {
                const int row = node_local * 32 + mt * 16 + l16;
                const uint4* pa =
                    reinterpret_cast<const uint4*>(&s_h1[row * HDIM + c0]);
                a[mt].q[0] = pa[0];   // k = c0..c0+7
                a[mt].q[1] = pa[2];   // k = c0+16..c0+23
            }
            const uint32_t* base = &s_w2[((ks * 16 + nq * 4) * 32 + lane) * 8];
            OpPack b[4];
#pragma unroll
            for (int nt2 = 0; nt2 < 4; ++nt2) {
                const uint4* pb = reinterpret_cast<const uint4*>(base + nt2 * 256);
                b[nt2].q[0] = pb[0];
                b[nt2].q[1] = pb[1];
            }
#pragma unroll
            for (int nt2 = 0; nt2 < 4; ++nt2) {
                acc2[0][nt2] = __builtin_amdgcn_wmma_f32_16x16x32_bf16(
                    false, a[0].v, false, b[nt2].v, (short)0, acc2[0][nt2], false, false);
                acc2[1][nt2] = __builtin_amdgcn_wmma_f32_16x16x32_bf16(
                    false, a[1].v, false, b[nt2].v, (short)0, acc2[1][nt2], false, false);
            }
        }

        // epilogue: max over the node's 32 edges + bias, write fp32 output
#pragma unroll
        for (int nt2 = 0; nt2 < 4; ++nt2) {
            float m = -3.402823466e+38f;
#pragma unroll
            for (int mt = 0; mt < 2; ++mt)
#pragma unroll
                for (int r = 0; r < 8; ++r)
                    m = fmaxf(m, acc2[mt][nt2][r]);
            m = fmaxf(m, __shfl_xor(m, 16, 32));   // combine lane halves (rows)
            m += bia2[nt2];
            if (lane < 16)
                out[(size_t)gnode * HDIM + nq * 64 + nt2 * 16 + l16] = m;
        }
        __syncthreads();   // protect s_h1 before next group's writes
    }
}

extern "C" void kernel_launch(void* const* d_in, const int* in_sizes, int n_in,
                              void* d_out, int out_size, void* d_ws, size_t ws_size,
                              hipStream_t stream) {
    const float* X       = (const float*)d_in[0];   // node_features [20000,128]
    const int*   senders = (const int*)d_in[1];     // [E]
    // d_in[2] = receivers (implicit repeat pattern; unused)
    const float* W1 = (const float*)d_in[3];        // [256,256]
    const float* b1 = (const float*)d_in[4];        // [256]
    const float* W2 = (const float*)d_in[5];        // [256,256]
    const float* b2 = (const float*)d_in[6];        // [256]
    float* out = (float*)d_out;                     // [20000,256]

    uint32_t* wsw = (uint32_t*)d_ws;                // 256 KB swizzled bf16 weights

    prep_weights_k<<<128, 256, 0, stream>>>(W1, W2, wsw);
    edgeconv_main_k<<<1024, 256, 0, stream>>>(X, senders, b1, b2, wsw, out);
}
